// DiffHead_58231166599757
// MI455X (gfx1250) — matrix operand order repaired and checked by reading, not compiled, not run on hardware
//
#include <hip/hip_runtime.h>
#include <hip/hip_bf16.h>

typedef __attribute__((ext_vector_type(16))) _Float16 v16h;
typedef __attribute__((ext_vector_type(8)))  float    v8f;
typedef int v4i_ __attribute__((vector_size(16)));
typedef __attribute__((address_space(1))) v4i_ gv4i;   // global (prints as __device__)
typedef __attribute__((address_space(3))) v4i_ lv4i;   // LDS

#define HS     128
#define SEQ    2048
#define CDIM   1024
#define BATCH  8

#if __has_builtin(__builtin_amdgcn_global_load_async_to_lds_b128) && \
    __has_builtin(__builtin_amdgcn_s_wait_asynccnt)
#define HAVE_ASYNC_LDS 1
#else
#define HAVE_ASYNC_LDS 0
#endif

#if HAVE_ASYNC_LDS
static __device__ __forceinline__ void async_copy16(const void* gsrc, void* ldst)
{
    __builtin_amdgcn_global_load_async_to_lds_b128((gv4i*)gsrc, (lv4i*)ldst, 0, 0);
}
#endif

// ---------------------------------------------------------------------------
// Kernel 0: fp32 -> f16 weight conversion (tiny, runs once)
// ---------------------------------------------------------------------------
__global__ __launch_bounds__(256) void cvt_f32_to_f16(const float* __restrict__ src,
                                                      _Float16* __restrict__ dst, int n)
{
    int i = blockIdx.x * 256 + threadIdx.x;
    if (i < n) dst[i] = (_Float16)src[i];
}

// ---------------------------------------------------------------------------
// Kernel 1: Ch[M,N] = (f16)(A_f32[M,1024] @ Wh_f16[1024,N])
// One block per 16-row tile: A panel staged once into LDS as f16 (A is read
// exactly once from HBM). 8 waves cover the N/16 column tiles, A-frag reused.
// ---------------------------------------------------------------------------
template <int N>
__global__ __launch_bounds__(256) void proj_gemm_v2(
    const float* __restrict__ A, const _Float16* __restrict__ Wh,
    _Float16* __restrict__ Ch)
{
    __shared__ _Float16 As[16 * CDIM];          // 32 KB

    const int tid  = threadIdx.x;
    const int wave = tid >> 5, lane = tid & 31;
    const int l = lane & 15, hi = lane >> 4;
    const int mt = blockIdx.x;

    {
        const float2* src = (const float2*)(A + (size_t)mt * 16 * CDIM);
        unsigned int* dst = (unsigned int*)As;
        for (int idx = tid; idx < 16 * CDIM / 2; idx += 256) {
            float2 f = src[idx];
            union { unsigned int u; _Float16 h[2]; } pk;
            pk.h[0] = (_Float16)f.x;
            pk.h[1] = (_Float16)f.y;
            dst[idx] = pk.u;
        }
    }
    __syncthreads();

    constexpr int NT = N >> 4;                  // 16 (QK) or 8 (V) column tiles
    const int nt0 = wave & (NT - 1);
    const _Float16* arow = As + l * CDIM;

    v8f acc0 = {}, acc1 = {};
    for (int kc = 0; kc < CDIM; kc += 32) {
        v16h a;
#pragma unroll
        for (int v = 0; v < 8; ++v) {
            int ka = kc + 2 * v + 16 * hi;                        // A layout
            a[2 * v]     = arow[ka];
            a[2 * v + 1] = arow[ka + 1];
        }
        v16h b0;
#pragma unroll
        for (int v = 0; v < 8; ++v) {
            int kk = kc + 2 * v + 16 * hi;                        // B layout
            b0[2 * v]     = Wh[(size_t)kk * N + nt0 * 16 + l];
            b0[2 * v + 1] = Wh[(size_t)(kk + 1) * N + nt0 * 16 + l];
        }
        acc0 = __builtin_amdgcn_wmma_f32_16x16x32_f16(false, a, false, b0,
                                                      (short)0, acc0, false, false);
        if (NT == 16) {
            v16h b1;
#pragma unroll
            for (int v = 0; v < 8; ++v) {
                int kk = kc + 2 * v + 16 * hi;
                b1[2 * v]     = Wh[(size_t)kk * N + (nt0 + 8) * 16 + l];
                b1[2 * v + 1] = Wh[(size_t)(kk + 1) * N + (nt0 + 8) * 16 + l];
            }
            acc1 = __builtin_amdgcn_wmma_f32_16x16x32_f16(false, a, false, b1,
                                                          (short)0, acc1, false, false);
        }
    }
#pragma unroll
    for (int r = 0; r < 8; ++r) {
        Ch[(size_t)(mt * 16 + r + 8 * hi) * N + nt0 * 16 + l] = (_Float16)acc0[r];
        if (NT == 16)
            Ch[(size_t)(mt * 16 + r + 8 * hi) * N + (nt0 + 8) * 16 + l] = (_Float16)acc1[r];
    }
}

// ---------------------------------------------------------------------------
// Attention helpers (A/B operands use contiguous k order, k = 16*hi + s)
// ---------------------------------------------------------------------------
static __device__ __forceinline__ v8f sQK(const _Float16* Qs, const _Float16* Ks,
                                          int wq, int l, int hi, int ksub)
{
    v8f acc = {};
    const _Float16* qrow = Qs + (wq * 16 + l) * HS;
    const _Float16* krow = Ks + (ksub * 16 + l) * HS;
#pragma unroll
    for (int kc = 0; kc < HS; kc += 32) {
        v16h a, b;
#pragma unroll
        for (int v = 0; v < 8; ++v) {
            int ka = kc + 2 * v + 16 * hi;
            a[2 * v]     = qrow[ka];
            a[2 * v + 1] = qrow[ka + 1];
            int kk = kc + 2 * v + 16 * hi;
            b[2 * v]     = krow[kk];
            b[2 * v + 1] = krow[kk + 1];
        }
        acc = __builtin_amdgcn_wmma_f32_16x16x32_f16(false, a, false, b,
                                                     (short)0, acc, false, false);
    }
    return acc;
}

static __device__ __forceinline__ void online_update(
    v8f Sa, v8f Sb, const int* qrow, const int* padr, int kbase, int l, int hi,
    float scl, float fill_logit, float* m, float* z, v8f* O, _Float16* Pw)
{
    const int keyA = kbase + l;
    const int keyB = kbase + 16 + l;
#pragma unroll
    for (int r = 0; r < 8; ++r) {
        bool fa = (keyA > qrow[r] + 1) || padr[r];
        bool fb = (keyB > qrow[r] + 1) || padr[r];
        float la = fa ? fill_logit : Sa[r] * scl;
        float lb = fb ? fill_logit : Sb[r] * scl;
        float mx = fmaxf(la, lb);
#pragma unroll
        for (int msk = 8; msk >= 1; msk >>= 1) mx = fmaxf(mx, __shfl_xor(mx, msk, 32));
        float mn   = fmaxf(m[r], mx);
        float corr = __expf(m[r] - mn);
        float ea   = __expf(la - mn);
        float eb   = __expf(lb - mn);
        float rs   = ea + eb;
#pragma unroll
        for (int msk = 8; msk >= 1; msk >>= 1) rs += __shfl_xor(rs, msk, 32);
        z[r] = z[r] * corr + rs;
        m[r] = mn;
#pragma unroll
        for (int nt = 0; nt < 8; ++nt) O[nt][r] *= corr;
        Pw[(r + 8 * hi) * 32 + l]      = (_Float16)ea;
        Pw[(r + 8 * hi) * 32 + 16 + l] = (_Float16)eb;
    }
}

static __device__ __forceinline__ void pv_accum(const _Float16* Pw, const _Float16* Vs,
                                                int l, int hi, v8f* O)
{
    v16h a;
    const _Float16* prow = Pw + l * 32;
#pragma unroll
    for (int v = 0; v < 8; ++v) {
        int ka = 2 * v + 16 * hi;
        a[2 * v]     = prow[ka];
        a[2 * v + 1] = prow[ka + 1];
    }
#pragma unroll
    for (int nt = 0; nt < 8; ++nt) {
        v16h bv;
#pragma unroll
        for (int v = 0; v < 8; ++v) {
            int kk = 2 * v + 16 * hi;
            bv[2 * v]     = Vs[kk * HS + nt * 16 + l];
            bv[2 * v + 1] = Vs[(kk + 1) * HS + nt * 16 + l];
        }
        O[nt] = __builtin_amdgcn_wmma_f32_16x16x32_f16(false, a, false, bv,
                                                       (short)0, O[nt], false, false);
    }
}

// ---------------------------------------------------------------------------
// Kernel 2: differential flash attention. 4 waves/block, 16 query rows/wave.
// ---------------------------------------------------------------------------
__global__ __launch_bounds__(128) void diff_attn_kernel(
    const _Float16* __restrict__ Qh, const _Float16* __restrict__ Kh,
    const _Float16* __restrict__ Vh, const int* __restrict__ pad,
    const float* __restrict__ lq1, const float* __restrict__ lk1,
    const float* __restrict__ lq2, const float* __restrict__ lk2,
    float* __restrict__ out)
{
    __shared__ _Float16 Qs1[64 * HS];
    __shared__ _Float16 Qs2[64 * HS];
    __shared__ _Float16 Ks1[32 * HS];
    __shared__ _Float16 Ks2[32 * HS];
    __shared__ _Float16 Vs [32 * HS];
    __shared__ _Float16 Pb [4 * 16 * 32];

    const int tid  = threadIdx.x;
    const int wq   = tid >> 5, lane = tid & 31;
    const int l    = lane & 15, hi = lane >> 4;
    const int b    = blockIdx.x >> 5;
    const int qblk = blockIdx.x & 31;
    const int qbase = qblk * 64;

    float d1 = 0.f, d2 = 0.f;
#pragma unroll
    for (int i = 0; i < 4; ++i) {
        int idx = lane * 4 + i;
        d1 += lq1[idx] * lk1[idx];
        d2 += lq2[idx] * lk2[idx];
    }
#pragma unroll
    for (int msk = 16; msk >= 1; msk >>= 1) {
        d1 += __shfl_xor(d1, msk, 32);
        d2 += __shfl_xor(d2, msk, 32);
    }
    const float lbd = __expf(d1) - __expf(d2) + 0.8f;

    // stage this block's 64 query rows (Q1 | Q2 split of the 256-wide rows)
#if HAVE_ASYNC_LDS
    {
        const _Float16* qg = Qh + ((size_t)b * SEQ + qbase) * (2 * HS);
        for (int idx = tid; idx < 2048; idx += 128) {     // 64 rows x 32 chunks of 16B
            int row = idx >> 5, c = idx & 31;
            const _Float16* src = qg + row * 256 + c * 8;
            _Float16* dst = (c < 16 ? Qs1 : Qs2) + row * 128 + (c & 15) * 8;
            async_copy16(src, dst);
        }
    }
#else
    {
        const unsigned int* qg = (const unsigned int*)(Qh + ((size_t)b * SEQ + qbase) * (2 * HS));
        unsigned int* s1 = (unsigned int*)Qs1;
        unsigned int* s2 = (unsigned int*)Qs2;
        for (int idx = tid; idx < 64 * 64; idx += 128) {
            int r = idx >> 6, cu = idx & 63;
            s1[idx] = qg[(size_t)r * 128 + cu];
            s2[idx] = qg[(size_t)r * 128 + 64 + cu];
        }
    }
#endif

    int qrow[8], padr[8];
#pragma unroll
    for (int r = 0; r < 8; ++r) {
        qrow[r] = qbase + wq * 16 + r + 8 * hi;
        padr[r] = pad[(size_t)b * SEQ + qrow[r]];
    }

    const float scl        = 0.08838834764831845f;   // 128^-0.5
    const float fill_logit = 1e-9f * scl;

    float m1[8], z1[8], m2[8], z2[8];
    v8f O1[8], O2[8];
    const v8f vzero = {};
#pragma unroll
    for (int r = 0; r < 8; ++r) { m1[r] = -3.0e38f; z1[r] = 0.f; m2[r] = -3.0e38f; z2[r] = 0.f; }
#pragma unroll
    for (int nt = 0; nt < 8; ++nt) { O1[nt] = vzero; O2[nt] = vzero; }

    _Float16* Pw = Pb + wq * (16 * 32);

    for (int kb = 0; kb < SEQ / 32; ++kb) {
        const int kbase = kb * 32;
        __syncthreads();
#if HAVE_ASYNC_LDS
        {
            const _Float16* kg = Kh + ((size_t)b * SEQ + kbase) * (2 * HS);
            const _Float16* vg = Vh + ((size_t)b * SEQ + kbase) * HS;
            for (int idx = tid; idx < 1536; idx += 128) {  // K1|K2 16KB + V 8KB in 16B chunks
                const _Float16* src;
                _Float16* dst;
                if (idx < 1024) {
                    int row = idx >> 5, c = idx & 31;
                    src = kg + row * 256 + c * 8;
                    dst = (c < 16 ? Ks1 : Ks2) + row * 128 + (c & 15) * 8;
                } else {
                    int j = idx - 1024;
                    src = vg + j * 8;
                    dst = Vs + j * 8;
                }
                async_copy16(src, dst);
            }
            __builtin_amdgcn_s_wait_asynccnt(0);
        }
#else
        {
            const unsigned int* kg = (const unsigned int*)(Kh + ((size_t)b * SEQ + kbase) * (2 * HS));
            const unsigned int* vg = (const unsigned int*)(Vh + ((size_t)b * SEQ + kbase) * HS);
            unsigned int* s1 = (unsigned int*)Ks1;
            unsigned int* s2 = (unsigned int*)Ks2;
            unsigned int* sv = (unsigned int*)Vs;
            for (int idx = tid; idx < 32 * 64; idx += 128) {
                int r = idx >> 6, cu = idx & 63;
                s1[idx] = kg[(size_t)r * 128 + cu];
                s2[idx] = kg[(size_t)r * 128 + 64 + cu];
                sv[idx] = vg[idx];
            }
        }
#endif
        if (kb + 1 < SEQ / 32)
            __builtin_prefetch((const void*)(Kh + ((size_t)b * SEQ + kbase + 32) * (2 * HS)), 0, 1);
        __syncthreads();

        // ---- s1 path ----
        {
            v8f Sa = sQK(Qs1, Ks1, wq, l, hi, 0);
            v8f Sb = sQK(Qs1, Ks1, wq, l, hi, 1);
            online_update(Sa, Sb, qrow, padr, kbase, l, hi, scl, fill_logit, m1, z1, O1, Pw);
        }
        __syncthreads();
        pv_accum(Pw, Vs, l, hi, O1);
        __syncthreads();

        // ---- s2 path ----
        {
            v8f Sa = sQK(Qs2, Ks2, wq, l, hi, 0);
            v8f Sb = sQK(Qs2, Ks2, wq, l, hi, 1);
            online_update(Sa, Sb, qrow, padr, kbase, l, hi, scl, fill_logit, m2, z2, O2, Pw);
        }
        __syncthreads();
        pv_accum(Pw, Vs, l, hi, O2);
    }

#pragma unroll
    for (int nt = 0; nt < 8; ++nt) {
#pragma unroll
        for (int r = 0; r < 8; ++r) {
            float o = O1[nt][r] / z1[r] - lbd * (O2[nt][r] / z2[r]);
            out[((size_t)b * SEQ + qrow[r]) * HS + nt * 16 + l] = o;
        }
    }
}

// ---------------------------------------------------------------------------
extern "C" void kernel_launch(void* const* d_in, const int* in_sizes, int n_in,
                              void* d_out, int out_size, void* d_ws, size_t ws_size,
                              hipStream_t stream)
{
    (void)in_sizes; (void)n_in; (void)out_size; (void)ws_size;
    const float* q   = (const float*)d_in[0];
    const float* k   = (const float*)d_in[1];
    const float* v   = (const float*)d_in[2];
    const int*   pad = (const int*)  d_in[3];
    const float* Wq  = (const float*)d_in[4];
    const float* Wk  = (const float*)d_in[5];
    const float* Wv  = (const float*)d_in[6];
    const float* lq1 = (const float*)d_in[7];
    const float* lk1 = (const float*)d_in[8];
    const float* lq2 = (const float*)d_in[9];
    const float* lk2 = (const float*)d_in[10];
    float* out = (float*)d_out;

    // workspace (f16): Qh 8MB | Kh 8MB | Vh 4MB | Wqh .5MB | Wkh .5MB | Wvh .25MB
    _Float16* Qh  = (_Float16*)d_ws;
    _Float16* Kh  = Qh  + (size_t)BATCH * SEQ * 2 * HS;
    _Float16* Vh  = Kh  + (size_t)BATCH * SEQ * 2 * HS;
    _Float16* Wqh = Vh  + (size_t)BATCH * SEQ * HS;
    _Float16* Wkh = Wqh + (size_t)CDIM * 256;
    _Float16* Wvh = Wkh + (size_t)CDIM * 256;

    cvt_f32_to_f16<<<dim3(CDIM * 256 / 256), dim3(256), 0, stream>>>(Wq, Wqh, CDIM * 256);
    cvt_f32_to_f16<<<dim3(CDIM * 256 / 256), dim3(256), 0, stream>>>(Wk, Wkh, CDIM * 256);
    cvt_f32_to_f16<<<dim3(CDIM * 128 / 256), dim3(256), 0, stream>>>(Wv, Wvh, CDIM * 128);

    const int MT = BATCH * SEQ / 16;   // 1024 row tiles
    proj_gemm_v2<256><<<dim3(MT), dim3(256), 0, stream>>>(q, Wqh, Qh);
    proj_gemm_v2<256><<<dim3(MT), dim3(256), 0, stream>>>(k, Wkh, Kh);
    proj_gemm_v2<128><<<dim3(MT), dim3(256), 0, stream>>>(v, Wvh, Vh);

    diff_attn_kernel<<<dim3(BATCH * (SEQ / 64)), dim3(128), 0, stream>>>(
        Qh, Kh, Vh, pad, lq1, lk1, lq2, lk2, out);
}
